// MultiHeadAttention_35570919145591
// MI455X (gfx1250) — matrix operand order repaired and checked by reading, not compiled, not run on hardware
//
#include <hip/hip_runtime.h>
#include <hip/hip_bf16.h>

// MI455X / gfx1250 MHA forward. Compute-bound (~200 GFLOP) -> all three matmul
// stages go through v_wmma_f32_16x16x32_bf16 (fp32 accumulate). GEMMs use the
// CDNA5 async-to-LDS path (ASYNCcnt) with double buffering so WMMAs overlap
// the tile DMA. Wave32 everywhere.

typedef __bf16 bf16;
typedef __attribute__((ext_vector_type(16))) __bf16 v16bf;
typedef __attribute__((ext_vector_type(8)))  __bf16 v8bf;
typedef __attribute__((ext_vector_type(8)))  float  v8f;

#define BATCH 2
#define TSEQ  2048
#define CDIM  2048
#define HEADS 16
#define DHEAD 128

#define BM 128
#define BN 128
#define BK 32

__device__ __forceinline__ bf16 f2bf(float f) {
  unsigned u = __builtin_bit_cast(unsigned, f);
  unsigned r = u + 0x7FFFu + ((u >> 16) & 1u);   // round-to-nearest-even
  unsigned short h = (unsigned short)(r >> 16);
  return __builtin_bit_cast(bf16, h);
}

__device__ __forceinline__ v16bf cat16(v8bf lo, v8bf hi) {
  return __builtin_shufflevector(lo, hi, 0,1,2,3,4,5,6,7,8,9,10,11,12,13,14,15);
}

// A-matrix 16x32 bf16 fragment (row-major source, leading dim ld).
// Lane L<16: M=L, K = {0..7} elems 0..7, {16..23} elems 8..15.
// Lane L>=16: M=L-16, K = {8..15}, {24..31}.
__device__ __forceinline__ v16bf load_a_frag(const bf16* base, int ld, int lane) {
  int m = lane & 15, hi = (lane >> 4) & 1;
  const bf16* p = base + (size_t)m * ld + hi * 8;
  v8bf lo8 = *(const v8bf*)p;
  v8bf hi8 = *(const v8bf*)(p + 16);
  return cat16(lo8, hi8);
}

// B-matrix 32x16 bf16 fragment from column-major-stored B (Bt[n][k], ld):
// lane<16: N=lane, elems = K {0..7},{16..23}; lane>=16: N=lane-16, elems = K {8..15},{24..31}.
__device__ __forceinline__ v16bf load_b_frag(const bf16* base, int ld, int lane) {
  int n = lane & 15, hi = (lane >> 4) & 1;
  const bf16* p = base + (size_t)n * ld + hi * 8;
  v8bf lo8 = *(const v8bf*)p;
  v8bf hi8 = *(const v8bf*)(p + 16);
  return cat16(lo8, hi8);
}

__device__ __forceinline__ v8f wmma_bf16f32(v16bf a, v16bf b, v8f c) {
  return __builtin_amdgcn_wmma_f32_16x16x32_bf16(false, a, false, b, (short)0, c,
                                                 false, false);
}

// CDNA5 async copy: global -> LDS, tracked with ASYNCcnt. Inline asm keeps it
// portable across ROCm-7.2 / amdgpu-toolchain builtin arity differences.
// %0 = VGPR holding workgroup-relative LDS byte address, %1 = 64-bit global addr.
__device__ __forceinline__ void async_load_b128(unsigned lds_off, const void* gptr) {
  asm volatile("global_load_async_to_lds_b128 %0, %1, off"
               :: "v"(lds_off), "v"((unsigned long long)(size_t)gptr)
               : "memory");
}

__device__ __forceinline__ float rmax16(float v) {
#pragma unroll
  for (int m = 1; m < 16; m <<= 1) v = fmaxf(v, __shfl_xor(v, m, 16));
  return v;
}
__device__ __forceinline__ float rsum16(float v) {
#pragma unroll
  for (int m = 1; m < 16; m <<= 1) v += __shfl_xor(v, m, 16);
  return v;
}

// ---------------------------------------------------------------- casts
__global__ void cast_kernel(const float* __restrict__ s, bf16* __restrict__ d, int n) {
  int i = blockIdx.x * blockDim.x + threadIdx.x;
  if (i < n) d[i] = f2bf(s[i]);
}

// src[K][N] f32 -> dst[N][K] bf16 (weights pre-transposed for contiguous B-frags)
__global__ void transpose_cast_kernel(const float* __restrict__ src,
                                      bf16* __restrict__ dst, int K, int N) {
  int idx = blockIdx.x * blockDim.x + threadIdx.x;
  if (idx >= K * N) return;
  int n = idx / K, k = idx - n * K;
  dst[idx] = f2bf(src[(size_t)k * N + n]);
}

// ---------------------------------------------------------------- GEMM
// C[M,N] = A[M,K] (bf16 row-major) @ B (bf16 stored transposed Bt[N][K]).
// Block = 256 threads (8 waves) on a 128x128 C tile. Per 32-k step, A/B tiles
// (8 KB each) are DMA'd to LDS with global_load_async_to_lds_b128, double
// buffered; each wave computes a 32x64 sub-tile (8 WMMAs) out of LDS.
// mode 0: f32 store. mode 1: fused qkv scatter (q,k -> f32 [BH][T][D],
// v -> bf16 transposed [BH][D][T]).
__global__ void __launch_bounds__(256)
gemm_wmma(const bf16* __restrict__ A, const bf16* __restrict__ Bt,
          float* __restrict__ C, int M, int N, int K, int mode,
          float* __restrict__ qf, float* __restrict__ kf, bf16* __restrict__ vbt) {
  __shared__ bf16 sA[2][BM * BK];   // [row][k] row-major
  __shared__ bf16 sB[2][BN * BK];   // [n][k]

  int tid = threadIdx.x;
  int lane = tid & 31, wv = tid >> 5;
  int rg = wv >> 1, cg = wv & 1;            // 4 row-groups x 2 col-groups

  int ntn = N / BN;
  int tm = blockIdx.x / ntn, tn = blockIdx.x - tm * ntn;

  const bf16* Ag = A + (size_t)tm * BM * K;
  const bf16* Bg = Bt + (size_t)tn * BN * K;

  // 512 16-byte chunks per 8KB tile; 2 chunks per thread per tile.
  // chunk ch: row = ch>>2, k-chunk = ch&3; LDS offset = ch*16 bytes.
  auto issue = [&](int buf, int k0) {
    unsigned baseA = (unsigned)(size_t)&sA[buf][0];
    unsigned baseB = (unsigned)(size_t)&sB[buf][0];
#pragma unroll
    for (int c = 0; c < 2; c++) {
      int ch = tid * 2 + c;
      int row = ch >> 2, kc = ch & 3;
      async_load_b128(baseA + ch * 16, Ag + (size_t)row * K + k0 + kc * 8);
      async_load_b128(baseB + ch * 16, Bg + (size_t)row * K + k0 + kc * 8);
    }
  };

  v8f acc[2][4];
#pragma unroll
  for (int i = 0; i < 2; i++)
#pragma unroll
    for (int j = 0; j < 4; j++) { v8f z = {0,0,0,0,0,0,0,0}; acc[i][j] = z; }

  int nk = K / BK;
  issue(0, 0);
  for (int it = 0; it < nk; ++it) {
    int buf = it & 1;
    if (it + 1 < nk) {
      issue(buf ^ 1, (it + 1) * BK);
      // 4 loads outstanding for next buffer; <=4 means current buffer landed.
      asm volatile("s_wait_asynccnt 0x4" ::: "memory");
    } else {
      asm volatile("s_wait_asynccnt 0x0" ::: "memory");
    }
    __syncthreads();                         // current buffer visible to all waves

    v16bf a0 = load_a_frag(&sA[buf][(rg * 32) * BK], BK, lane);
    v16bf a1 = load_a_frag(&sA[buf][(rg * 32 + 16) * BK], BK, lane);
#pragma unroll
    for (int j = 0; j < 4; j++) {
      v16bf b = load_b_frag(&sB[buf][(cg * 64 + j * 16) * BK], BK, lane);
      acc[0][j] = wmma_bf16f32(a0, b, acc[0][j]);
      acc[1][j] = wmma_bf16f32(a1, b, acc[1][j]);
    }
    __syncthreads();                         // all reads done before buf is re-filled
  }

  int n = lane & 15, hi = (lane >> 4) & 1;
#pragma unroll
  for (int i = 0; i < 2; i++)
#pragma unroll
    for (int j = 0; j < 4; j++)
#pragma unroll
      for (int r = 0; r < 8; r++) {
        int row = tm * BM + rg * 32 + i * 16 + r + hi * 8;
        int col = tn * BN + cg * 64 + j * 16 + n;
        float v = acc[i][j][r];
        if (mode == 0) {
          C[(size_t)row * N + col] = v;
        } else {
          int b = row >> 11;            // row / TSEQ
          int t = row & (TSEQ - 1);
          int s = col >> 11;            // 0:q 1:k 2:v
          int h = (col >> 7) & (HEADS - 1);
          int d = col & (DHEAD - 1);
          int bh = b * HEADS + h;
          if (s == 0)      qf[((size_t)bh * TSEQ + t) * DHEAD + d] = v;
          else if (s == 1) kf[((size_t)bh * TSEQ + t) * DHEAD + d] = v;
          else             vbt[((size_t)bh * DHEAD + d) * TSEQ + t] = f2bf(v);
        }
      }
}

// ---------------------------------------------------------------- RoPE
__global__ void rope_kernel(const float* __restrict__ qf, const float* __restrict__ kf,
                            const float* __restrict__ cs, const float* __restrict__ sn,
                            bf16* __restrict__ qb, bf16* __restrict__ kb) {
  int idx = blockIdx.x * blockDim.x + threadIdx.x;   // [0, BH*T*64)
  int j  = idx & 63;
  int t  = (idx >> 6) & (TSEQ - 1);
  int bh = idx >> 17;                                 // / (T*64)
  if (bh >= BATCH * HEADS) return;
  size_t base = ((size_t)bh * TSEQ + t) * DHEAD;
  float c = cs[t * 64 + j], s = sn[t * 64 + j];
  float q1 = qf[base + j], q2 = qf[base + 64 + j];
  float k1 = kf[base + j], k2 = kf[base + 64 + j];
  qb[base + j]      = f2bf(q1 * c - q2 * s);
  qb[base + 64 + j] = f2bf(q2 * c + q1 * s);
  kb[base + j]      = f2bf(k1 * c - k2 * s);
  kb[base + 64 + j] = f2bf(k2 * c + k1 * s);
}

// ---------------------------------------------------------------- attention
// One wave owns one (bh, 16-query) tile; 4 independent waves per block.
// Per 32-key step: 8 WMMAs for S = Q K^T, online softmax (16-lane shfl
// reductions, wave32), P transposed C->A layout through LDS with an explicit
// s_wait_dscnt, then 8 WMMAs for O += P V.
__global__ void __launch_bounds__(128)
attn_kernel(const bf16* __restrict__ qb, const bf16* __restrict__ kb,
            const bf16* __restrict__ vbt, bf16* __restrict__ attn) {
  __shared__ bf16 pb[4][16][32];
  int lane = threadIdx.x & 31;
  int warp = threadIdx.x >> 5;
  int tile = blockIdx.x * 4 + warp;       // BH * (T/16) tiles
  int qt = tile & (TSEQ / 16 - 1);
  int bh = tile >> 7;                      // / (T/16)
  int q0 = qt * 16;
  int n = lane & 15, hi = (lane >> 4) & 1;
  const float scale = 0.0883883476483184f; // 1/sqrt(128)

  const bf16* qbase = qb + ((size_t)bh * TSEQ + q0) * DHEAD;
  v16bf qa[4];
#pragma unroll
  for (int kk = 0; kk < 4; kk++) qa[kk] = load_a_frag(qbase + kk * 32, DHEAD, lane);

  v8f o[8];
  float mrow[8], lrow[8];
#pragma unroll
  for (int g = 0; g < 8; g++) { v8f z = {0,0,0,0,0,0,0,0}; o[g] = z; }
#pragma unroll
  for (int r = 0; r < 8; r++) { mrow[r] = -1e30f; lrow[r] = 0.0f; }

  int kend = q0 + 16;
  for (int ks = 0; ks < kend; ks += 32) {
    v8f s0 = {0,0,0,0,0,0,0,0}, s1 = {0,0,0,0,0,0,0,0};
#pragma unroll
    for (int kk = 0; kk < 4; kk++) {
      // K^T as B-matrix: lane = key column, elems = contiguous d.
      v16bf b0 = load_b_frag(kb + ((size_t)bh * TSEQ + ks) * DHEAD + kk * 32, DHEAD, lane);
      v16bf b1 = load_b_frag(kb + ((size_t)bh * TSEQ + ks + 16) * DHEAD + kk * 32, DHEAD, lane);
      s0 = wmma_bf16f32(qa[kk], b0, s0);
      s1 = wmma_bf16f32(qa[kk], b1, s1);
    }
#pragma unroll
    for (int r = 0; r < 8; r++) {
      int row = q0 + r + hi * 8;
      int key0 = ks + n, key1 = key0 + 16;
      float f0 = (key0 <= row) ? s0[r] * scale : -1e30f;
      float f1 = (key1 <= row) ? s1[r] * scale : -1e30f;
      float mx   = rmax16(fmaxf(f0, f1));
      float mnew = fmaxf(mrow[r], mx);
      float p0 = __expf(f0 - mnew);
      float p1 = __expf(f1 - mnew);
      float corr = __expf(mrow[r] - mnew);
      lrow[r] = lrow[r] * corr + rsum16(p0 + p1);
      mrow[r] = mnew;
#pragma unroll
      for (int g = 0; g < 8; g++) o[g][r] *= corr;
      int m = r + hi * 8;
      pb[warp][m][n]      = f2bf(p0);
      pb[warp][m][n + 16] = f2bf(p1);
    }
    // DS ops are wave-ordered; wait + clobber pins the store->load transpose.
    asm volatile("s_wait_dscnt 0x0" ::: "memory");
    const bf16* pp = &pb[warp][lane & 15][0];
    v8bf plo = *(const v8bf*)(pp + hi * 8);
    v8bf phi = *(const v8bf*)(pp + 16 + hi * 8);
    v16bf pa = cat16(plo, phi);
#pragma unroll
    for (int g = 0; g < 8; g++) {
      // V transposed [BH][D][T]: lane = d column, elems = contiguous keys.
      v16bf vf = load_b_frag(vbt + ((size_t)bh * DHEAD + g * 16) * TSEQ + ks, TSEQ, lane);
      o[g] = wmma_bf16f32(pa, vf, o[g]);
    }
  }

  int b = bh >> 4, h = bh & (HEADS - 1);
#pragma unroll
  for (int g = 0; g < 8; g++)
#pragma unroll
    for (int r = 0; r < 8; r++) {
      int row = q0 + r + hi * 8;
      float v = o[g][r] / lrow[r];
      attn[((size_t)(b * TSEQ + row)) * CDIM + h * DHEAD + g * 16 + n] = f2bf(v);
    }
}

// ---------------------------------------------------------------- launch
extern "C" void kernel_launch(void* const* d_in, const int* in_sizes, int n_in,
                              void* d_out, int out_size, void* d_ws, size_t ws_size,
                              hipStream_t stream) {
  const float* x    = (const float*)d_in[0];
  const float* wqkv = (const float*)d_in[1];
  const float* wout = (const float*)d_in[2];
  const float* cs   = (const float*)d_in[3];
  const float* sn   = (const float*)d_in[4];
  float* out = (float*)d_out;

  const int BT = BATCH * TSEQ;   // 4096
  const int BH = BATCH * HEADS;  // 32

  char* ws = (char*)d_ws;
  size_t off = 0;
  auto take = [&](size_t bytes) -> char* {
    char* p = ws + off;
    off = (off + bytes + 255) & ~(size_t)255;
    return p;
  };
  bf16*  xb    = (bf16*)take((size_t)BT * CDIM * 2);
  bf16*  wqkvT = (bf16*)take((size_t)3 * CDIM * CDIM * 2);
  bf16*  woutT = (bf16*)take((size_t)CDIM * CDIM * 2);
  float* qf    = (float*)take((size_t)BH * TSEQ * DHEAD * 4);
  float* kf    = (float*)take((size_t)BH * TSEQ * DHEAD * 4);
  bf16*  vbt   = (bf16*)take((size_t)BH * TSEQ * DHEAD * 2);
  bf16*  qb    = (bf16*)take((size_t)BH * TSEQ * DHEAD * 2);
  bf16*  kb    = (bf16*)take((size_t)BH * TSEQ * DHEAD * 2);
  bf16*  attn  = (bf16*)take((size_t)BT * CDIM * 2);

  cast_kernel<<<(BT * CDIM) / 256, 256, 0, stream>>>(x, xb, BT * CDIM);
  transpose_cast_kernel<<<(3 * CDIM * CDIM) / 256, 256, 0, stream>>>(wqkv, wqkvT,
                                                                     CDIM, 3 * CDIM);
  transpose_cast_kernel<<<(CDIM * CDIM) / 256, 256, 0, stream>>>(wout, woutT,
                                                                 CDIM, CDIM);
  {
    int blocks = (BT / BM) * ((3 * CDIM) / BN);   // 32 * 48 = 1536
    gemm_wmma<<<blocks, 256, 0, stream>>>(xb, wqkvT, nullptr, BT, 3 * CDIM, CDIM,
                                          1, qf, kf, vbt);
  }
  rope_kernel<<<(BH * TSEQ * 64) / 256, 256, 0, stream>>>(qf, kf, cs, sn, qb, kb);
  attn_kernel<<<(BH * (TSEQ / 16)) / 4, 128, 0, stream>>>(qb, kb, vbt, attn);
  {
    int blocks = (BT / BM) * (CDIM / BN);         // 32 * 16 = 512
    gemm_wmma<<<blocks, 256, 0, stream>>>(attn, woutT, out, BT, CDIM, CDIM,
                                          0, nullptr, nullptr, nullptr);
  }
}
